// GAT_88184268522161
// MI455X (gfx1250) — matrix operand order repaired and checked by reading, not compiled, not run on hardware
//
#include <hip/hip_runtime.h>

// ---------------------------------------------------------------------------
// 2-layer GAT for MI455X (gfx1250).  GEMMs via v_wmma_f32_16x16x32_bf16 with
// LDS-staged bf16 tiles (fully unrolled K loop), edge softmax via float
// atomics.  wave32 throughout.
// ---------------------------------------------------------------------------

typedef __attribute__((ext_vector_type(16))) __bf16 v16bf;
typedef __attribute__((ext_vector_type(8)))  float  v8f;

union Frag16 { v16bf v; unsigned int u[8]; };

__device__ __forceinline__ float leaky02(float x) { return x > 0.f ? x : 0.2f * x; }

// Float atomic max via sign-aware integer punning (correct for all finite floats).
__device__ __forceinline__ void atomicMaxF(float* addr, float v) {
    if (v >= 0.f) atomicMax((int*)addr, __float_as_int(v));
    else          atomicMin((unsigned int*)addr, __float_as_uint(v));
}

__global__ void fill_kernel(float* __restrict__ p, float v, int n) {
    int i = blockIdx.x * blockDim.x + threadIdx.x;
    if (i < n) p[i] = v;
}

// ---------------------------------------------------------------------------
// C[M,NC] = A[M,K] @ B[K,NC] (+ bias).  fp32 in/out, bf16 WMMA, f32 acc.
//
// Block = (NC/16) waves; each block computes a ROWS x NC macro-tile:
//   - A panel  [ROWS x K]  staged to LDS as bf16 (row-major)
//   - B matrix [K x NC]    staged to LDS as bf16 *transposed* -> [NC x K]
//     so every fragment read is K-contiguous => aligned dword ds_loads.
// Wave w owns column tile w and loops over ROWS/16 row tiles; K loop is
// fully unrolled (compile-time K) => K/32 back-to-back v_wmma per tile.
//
// WMMA fragment layouts used here (contiguous K, same for A and B):
//   A 16x32 bf16 : lanes 0-15 = rows; elems 0-15 -> K(half*16 .. +15).
//   B 32x16 bf16 : lane%16 = column; half-wave = K 0-15 / 16-31.
//   C/D v8f      : row = tile_row + half*8 + v, col = tile_col + lane%16.
// ---------------------------------------------------------------------------
template<int K, int NC, int ROWS>
__global__ void wmma_gemm_lds(const float* __restrict__ A,
                              const float* __restrict__ B,
                              const float* __restrict__ bias,
                              float* __restrict__ C, int M) {
    constexpr int NTHR = (NC / 16) * 32;
    __shared__ __bf16 sA[ROWS * K];     // [ROWS][K]
    __shared__ __bf16 sBt[NC * K];      // [NC][K]  (B transposed)

    const int tid  = threadIdx.x;
    const int wave = tid >> 5;
    const int lane = tid & 31;
    const int half = lane >> 4;
    const int l16  = lane & 15;
    const int rowB = blockIdx.x * ROWS;     // first row of this block's panel
    const int col0 = wave << 4;             // this wave's column tile

    // ---- stage A panel (clamp rows past M; read-only so clamping is safe) ----
    for (int i = tid; i < ROWS * K; i += NTHR) {
        const int r = i / K, k = i - r * K;
        const int rg = min(rowB + r, M - 1);
        sA[i] = (__bf16)A[(size_t)rg * K + k];
    }
    // ---- stage B transposed (coalesced global read over columns) ----
    for (int i = tid; i < K * NC; i += NTHR) {
        const int k = i / NC, c = i - k * NC;
        sBt[c * K + k] = (__bf16)B[i];
    }
    __syncthreads();

    const unsigned int* uA = (const unsigned int*)sA;
    const unsigned int* uB = (const unsigned int*)sBt;
    const int c    = col0 + l16;
    const float bv = bias ? bias[c] : 0.f;

    // B fragments depend only on (k0, lane): hoist dword bases once.
    const int bRow = (c * K) >> 1;

#pragma unroll
    for (int rt = 0; rt < ROWS / 16; ++rt) {
        v8f acc = {};
#pragma unroll
        for (int k0 = 0; k0 < K; k0 += 32) {
            Frag16 a, b;
            const int abase = ((rt * 16 + l16) * K + k0 + half * 16) >> 1;
#pragma unroll
            for (int j = 0; j < 4; ++j) {
                a.u[j]     = uA[abase + j];       // K = k0+half*16   + 2j,+1
                a.u[j + 4] = uA[abase + 4 + j];   // K = k0+half*16+8 + 2j,+1
            }
            const int bbase = bRow + ((k0 + half * 16) >> 1);
#pragma unroll
            for (int j = 0; j < 8; ++j) b.u[j] = uB[bbase + j];

            acc = __builtin_amdgcn_wmma_f32_16x16x32_bf16(
                      false, a.v, false, b.v, (short)0, acc, false, false);
        }
        // M % 16 == 0 => whole 16-row tile is in/out together (wave-uniform
        // branch, EXEC stays all-ones around the WMMAs).
        if (rowB + rt * 16 < M) {
#pragma unroll
            for (int v = 0; v < 8; ++v) {
                const int r = rowB + rt * 16 + half * 8 + v;
                C[(size_t)r * NC + c] = acc[v] + bv;
            }
        }
    }
}

// ------------------------- layer-1 attention coeffs -------------------------
// h1: [N,64] viewed as [N,8,8]; a_src/a_dst: [N,8]
__global__ void coeff1_kernel(const float* __restrict__ h1,
                              const float* __restrict__ att_s,
                              const float* __restrict__ att_d,
                              float* __restrict__ asrc, float* __restrict__ adst,
                              int N) {
    int t = blockIdx.x * blockDim.x + threadIdx.x;
    if (t >= N * 8) return;
    const int n = t >> 3, hd = t & 7;
    const float* hp = h1 + (size_t)n * 64 + hd * 8;
    float s = 0.f, d = 0.f;
#pragma unroll
    for (int c = 0; c < 8; ++c) {
        const float v = hp[c];
        s += v * att_s[hd * 8 + c];
        d += v * att_d[hd * 8 + c];
    }
    asrc[t] = s;
    adst[t] = d;
}

// ------------------------- layer-2 attention coeffs -------------------------
__global__ void coeff2_kernel(const float* __restrict__ h2,
                              const float* __restrict__ att_s,
                              const float* __restrict__ att_d,
                              float* __restrict__ asrc, float* __restrict__ adst,
                              int N) {
    int n = blockIdx.x * blockDim.x + threadIdx.x;
    if (n >= N) return;
    const float4* hp = (const float4*)(h2 + (size_t)n * 128);
    const float4* as = (const float4*)att_s;
    const float4* ad = (const float4*)att_d;
    float s = 0.f, d = 0.f;
#pragma unroll 4
    for (int i = 0; i < 32; ++i) {
        const float4 v = hp[i], a = as[i], b = ad[i];
        s += v.x * a.x + v.y * a.y + v.z * a.z + v.w * a.w;
        d += v.x * b.x + v.y * b.y + v.z * b.z + v.w * b.w;
    }
    asrc[n] = s;
    adst[n] = d;
}

__device__ __forceinline__ void edge_sd(const int* __restrict__ ei, int E, int e,
                                        int& s, int& d) {
    if (e < E) { s = ei[e]; d = ei[E + e]; }
    else       { s = e - E; d = e - E; }      // self-loops appended
}

// ------------------------- layer 1 edge passes (H=8) ------------------------
__global__ void edge_max1(const int* __restrict__ ei, int E, int Etot,
                          const float* __restrict__ asrc, const float* __restrict__ adst,
                          float* __restrict__ mx) {
    long long t = (long long)blockIdx.x * blockDim.x + threadIdx.x;
    if (t >= (long long)Etot * 8) return;
    const int e = (int)(t >> 3), hd = (int)(t & 7);
    int s, d; edge_sd(ei, E, e, s, d);
    const float l = leaky02(asrc[s * 8 + hd] + adst[d * 8 + hd]);
    atomicMaxF(&mx[d * 8 + hd], l);
}

__global__ void edge_sum1(const int* __restrict__ ei, int E, int Etot,
                          const float* __restrict__ asrc, const float* __restrict__ adst,
                          const float* __restrict__ mx, float* __restrict__ denom) {
    long long t = (long long)blockIdx.x * blockDim.x + threadIdx.x;
    if (t >= (long long)Etot * 8) return;
    const int e = (int)(t >> 3), hd = (int)(t & 7);
    int s, d; edge_sd(ei, E, e, s, d);
    const float l = leaky02(asrc[s * 8 + hd] + adst[d * 8 + hd]);
    atomicAdd(&denom[d * 8 + hd], __expf(l - mx[d * 8 + hd]));
}

__global__ void edge_scatter1(const int* __restrict__ ei, int E, int Etot,
                              const float* __restrict__ asrc, const float* __restrict__ adst,
                              const float* __restrict__ mx, const float* __restrict__ denom,
                              const float* __restrict__ h1, float* __restrict__ agg) {
    long long t = (long long)blockIdx.x * blockDim.x + threadIdx.x;
    if (t >= (long long)Etot * 8) return;
    const int e = (int)(t >> 3), hd = (int)(t & 7);
    int s, d; edge_sd(ei, E, e, s, d);
    const float l = leaky02(asrc[s * 8 + hd] + adst[d * 8 + hd]);
    const float alpha = __expf(l - mx[d * 8 + hd]) / denom[d * 8 + hd];
    const float4* hv = (const float4*)(h1 + (size_t)s * 64 + hd * 8);
    float* o = agg + (size_t)d * 64 + hd * 8;
    const float4 v0 = hv[0], v1 = hv[1];
    atomicAdd(o + 0, v0.x * alpha); atomicAdd(o + 1, v0.y * alpha);
    atomicAdd(o + 2, v0.z * alpha); atomicAdd(o + 3, v0.w * alpha);
    atomicAdd(o + 4, v1.x * alpha); atomicAdd(o + 5, v1.y * alpha);
    atomicAdd(o + 6, v1.z * alpha); atomicAdd(o + 7, v1.w * alpha);
}

// agg1 + b1 -> ELU -> h1e  ([N,64])
__global__ void elu_bias_kernel(const float* __restrict__ agg,
                                const float* __restrict__ b1,
                                float* __restrict__ out, int n) {
    int i = blockIdx.x * blockDim.x + threadIdx.x;
    if (i >= n) return;
    const float v = agg[i] + b1[i & 63];
    out[i] = v > 0.f ? v : (__expf(v) - 1.f);
}

// ------------------------- layer 2 edge passes (H=1, C=128) -----------------
__global__ void edge_max2(const int* __restrict__ ei, int E, int Etot,
                          const float* __restrict__ asrc, const float* __restrict__ adst,
                          float* __restrict__ mx) {
    long long t = (long long)blockIdx.x * blockDim.x + threadIdx.x;
    if (t >= Etot) return;
    const int e = (int)t;
    int s, d; edge_sd(ei, E, e, s, d);
    atomicMaxF(&mx[d], leaky02(asrc[s] + adst[d]));
}

__global__ void edge_sum2(const int* __restrict__ ei, int E, int Etot,
                          const float* __restrict__ asrc, const float* __restrict__ adst,
                          const float* __restrict__ mx, float* __restrict__ denom) {
    long long t = (long long)blockIdx.x * blockDim.x + threadIdx.x;
    if (t >= Etot) return;
    const int e = (int)t;
    int s, d; edge_sd(ei, E, e, s, d);
    const float l = leaky02(asrc[s] + adst[d]);
    atomicAdd(&denom[d], __expf(l - mx[d]));
}

// 32 threads per edge, 4 channels each (float4 gather, f32 atomic scatter)
__global__ void edge_scatter2(const int* __restrict__ ei, int E, int Etot,
                              const float* __restrict__ asrc, const float* __restrict__ adst,
                              const float* __restrict__ mx, const float* __restrict__ denom,
                              const float* __restrict__ h2, float* __restrict__ out) {
    long long t = (long long)blockIdx.x * blockDim.x + threadIdx.x;
    if (t >= (long long)Etot * 32) return;
    const int e = (int)(t >> 5), q = (int)(t & 31);
    int s, d; edge_sd(ei, E, e, s, d);
    const float l = leaky02(asrc[s] + adst[d]);
    const float alpha = __expf(l - mx[d]) / denom[d];
    const float4 v = ((const float4*)(h2 + (size_t)s * 128))[q];
    float* o = out + (size_t)d * 128 + q * 4;
    atomicAdd(o + 0, v.x * alpha); atomicAdd(o + 1, v.y * alpha);
    atomicAdd(o + 2, v.z * alpha); atomicAdd(o + 3, v.w * alpha);
}

__global__ void bias2_kernel(float* __restrict__ out, const float* __restrict__ b2, int n) {
    int i = blockIdx.x * blockDim.x + threadIdx.x;
    if (i < n) out[i] += b2[i & 127];
}

// ---------------------------------------------------------------------------
static inline int gridFor(long long work, int block) {
    return (int)((work + block - 1) / block);
}

extern "C" void kernel_launch(void* const* d_in, const int* in_sizes, int n_in,
                              void* d_out, int out_size, void* d_ws, size_t ws_size,
                              hipStream_t stream) {
    const float* x        = (const float*)d_in[0];
    const int*   ei       = (const int*)  d_in[1];
    const float* W_map    = (const float*)d_in[2];
    const float* b_map    = (const float*)d_in[3];
    const float* W1       = (const float*)d_in[4];
    const float* att_src1 = (const float*)d_in[5];
    const float* att_dst1 = (const float*)d_in[6];
    const float* b1       = (const float*)d_in[7];
    const float* W2       = (const float*)d_in[8];
    const float* att_src2 = (const float*)d_in[9];
    const float* att_dst2 = (const float*)d_in[10];
    const float* b2       = (const float*)d_in[11];

    const int N    = in_sizes[0] / 128;   // 50000, multiple of 16
    const int E    = in_sizes[1] / 2;     // 1.6M
    const int Etot = E + N;               // + self-loops

    // ---- workspace carve (floats) ----
    float* ws    = (float*)d_ws;
    float* h0    = ws; ws += (size_t)N * 128;   // reused as h1e after layer-1
    float* h1    = ws; ws += (size_t)N * 64;
    float* agg1  = ws; ws += (size_t)N * 64;
    float* h2    = ws; ws += (size_t)N * 128;
    float* asrc1 = ws; ws += (size_t)N * 8;
    float* adst1 = ws; ws += (size_t)N * 8;
    float* mx1   = ws; ws += (size_t)N * 8;
    float* dn1   = ws; ws += (size_t)N * 8;
    float* asrc2 = ws; ws += N;
    float* adst2 = ws; ws += N;
    float* mx2   = ws; ws += N;
    float* dn2   = ws; ws += N;
    float* out   = (float*)d_out;               // [N,128]

    const int BLK  = 256;
    const int ROWS = 64;                        // macro-tile rows per block

    // 1) h0 = x @ W_map + b_map          [N,128] (WMMA bf16, LDS-staged)
    wmma_gemm_lds<128, 128, 64><<<gridFor(N, ROWS), 256, 0, stream>>>(
        x, W_map, b_map, h0, N);
    // 2) h1 = h0 @ W1                     [N,64]
    wmma_gemm_lds<128, 64, 64><<<gridFor(N, ROWS), 128, 0, stream>>>(
        h0, W1, nullptr, h1, N);
    // 3) layer-1 attention coefficients
    coeff1_kernel<<<gridFor((long long)N * 8, BLK), BLK, 0, stream>>>(
        h1, att_src1, att_dst1, asrc1, adst1, N);
    // 4) init segment buffers
    fill_kernel<<<gridFor((long long)N * 8, BLK), BLK, 0, stream>>>(mx1, -1e30f, N * 8);
    fill_kernel<<<gridFor((long long)N * 8, BLK), BLK, 0, stream>>>(dn1, 0.f, N * 8);
    fill_kernel<<<gridFor((long long)N * 64, BLK), BLK, 0, stream>>>(agg1, 0.f, N * 64);
    // 5) softmax-aggregate over edges (per edge x head)
    const long long w1n = (long long)Etot * 8;
    edge_max1    <<<gridFor(w1n, BLK), BLK, 0, stream>>>(ei, E, Etot, asrc1, adst1, mx1);
    edge_sum1    <<<gridFor(w1n, BLK), BLK, 0, stream>>>(ei, E, Etot, asrc1, adst1, mx1, dn1);
    edge_scatter1<<<gridFor(w1n, BLK), BLK, 0, stream>>>(ei, E, Etot, asrc1, adst1, mx1, dn1, h1, agg1);
    // 6) h1e = elu(agg1 + b1)   (reuse h0 buffer)
    elu_bias_kernel<<<gridFor((long long)N * 64, BLK), BLK, 0, stream>>>(agg1, b1, h0, N * 64);
    // 7) h2 = h1e @ W2                    [N,128] (WMMA bf16, LDS-staged)
    wmma_gemm_lds<64, 128, 64><<<gridFor(N, ROWS), 256, 0, stream>>>(
        h0, W2, nullptr, h2, N);
    // 8) layer-2 attention coefficients
    coeff2_kernel<<<gridFor(N, BLK), BLK, 0, stream>>>(h2, att_src2, att_dst2, asrc2, adst2, N);
    // 9) init layer-2 segment buffers + output accumulator
    fill_kernel<<<gridFor(N, BLK), BLK, 0, stream>>>(mx2, -1e30f, N);
    fill_kernel<<<gridFor(N, BLK), BLK, 0, stream>>>(dn2, 0.f, N);
    fill_kernel<<<gridFor((long long)N * 128, BLK), BLK, 0, stream>>>(out, 0.f, N * 128);
    // 10) softmax-aggregate layer 2
    edge_max2<<<gridFor(Etot, BLK), BLK, 0, stream>>>(ei, E, Etot, asrc2, adst2, mx2);
    edge_sum2<<<gridFor(Etot, BLK), BLK, 0, stream>>>(ei, E, Etot, asrc2, adst2, mx2, dn2);
    edge_scatter2<<<gridFor((long long)Etot * 32, BLK), BLK, 0, stream>>>(
        ei, E, Etot, asrc2, adst2, mx2, dn2, h2, out);
    // 11) + b2
    bias2_kernel<<<gridFor((long long)N * 128, BLK), BLK, 0, stream>>>(out, b2, N * 128);
}
